// MultiHeadAttention_26723286515879
// MI455X (gfx1250) — compile-verified, matrix operand order checked by reading, not run on hardware
//
#include <hip/hip_runtime.h>
#include <cstdint>
#include <cstddef>

#define D_MODEL 768
#define NH 12
#define DK 64
#define BB 2
#define SS 2048
#define BS (BB * SS) /* 4096 */
#define KT 32        /* key tile for attention */

typedef __bf16 bf16;
typedef __attribute__((ext_vector_type(16))) __bf16 v16bf;
typedef __attribute__((ext_vector_type(2)))  __bf16 v2bf;
typedef __attribute__((ext_vector_type(8)))  float  v8f;
typedef __attribute__((ext_vector_type(4)))  int    v4i;

typedef __attribute__((address_space(1))) v4i* gv4i_p;   // global
typedef __attribute__((address_space(3))) v4i* lv4i_p;   // LDS

#if __has_builtin(__builtin_amdgcn_global_load_async_to_lds_b128)
#define HAVE_ASYNC_LDS 1
#else
#define HAVE_ASYNC_LDS 0
#endif

// ---------- bf16 conversion helpers (prefer v_cvt_pk_bf16_f32) ----------

__device__ __forceinline__ v2bf cvt2(float a, float b) {
#if __has_builtin(__builtin_amdgcn_cvt_pk_bf16_f32)
  auto t = __builtin_amdgcn_cvt_pk_bf16_f32(a, b);
  v2bf r; __builtin_memcpy(&r, &t, sizeof(r));
  return r;
#else
  v2bf r; r[0] = (bf16)a; r[1] = (bf16)b;
  return r;
#endif
}

__device__ __forceinline__ bf16 f2bf(float f) { return (bf16)f; }

__device__ __forceinline__ v8f wmma_bf16(v16bf a, v16bf b, v8f c) {
  return __builtin_amdgcn_wmma_f32_16x16x32_bf16(false, a, false, b, (short)0, c,
                                                 false, false);
}

__device__ __forceinline__ v16bf ld16(const bf16* p) { return *(const v16bf*)p; }

// 16-byte global -> LDS copy: async DMA if available, else load+ds_store
__device__ __forceinline__ void cp16_g2l(const bf16* g, bf16* l) {
#if HAVE_ASYNC_LDS
  __builtin_amdgcn_global_load_async_to_lds_b128(
      (gv4i_p)(void*)g, (lv4i_p)(void*)l, 0, 0);
#else
  *(uint4*)l = *(const uint4*)g;
#endif
}

__device__ __forceinline__ void stage_wait_and_sync() {
#if HAVE_ASYNC_LDS
#if __has_builtin(__builtin_amdgcn_s_wait_asynccnt)
  __builtin_amdgcn_s_wait_asynccnt(0);
#else
  asm volatile("s_wait_asynccnt 0x0" ::: "memory");
#endif
#endif
  __syncthreads();
}

// reductions across a 16-lane half-group (rows of a C fragment)
__device__ __forceinline__ float rmax16(float v) {
  v = fmaxf(v, __shfl_xor(v, 1, 16));
  v = fmaxf(v, __shfl_xor(v, 2, 16));
  v = fmaxf(v, __shfl_xor(v, 4, 16));
  v = fmaxf(v, __shfl_xor(v, 8, 16));
  return v;
}
__device__ __forceinline__ float rsum16(float v) {
  v += __shfl_xor(v, 1, 16);
  v += __shfl_xor(v, 2, 16);
  v += __shfl_xor(v, 4, 16);
  v += __shfl_xor(v, 8, 16);
  return v;
}

// ---------- kernel 0: fp32 -> bf16 one-pass conversion ----------

__global__ __launch_bounds__(256) void k_cvt(const float* __restrict__ src,
                                             bf16* __restrict__ dst, int n) {
  int base = (blockIdx.x * 256 + threadIdx.x) * 8;
  if (base >= n) return;
  const float4* q = (const float4*)(src + base);
  float4 a = q[0], b = q[1];
  union { v2bf p[4]; uint4 u; } o;
  o.p[0] = cvt2(a.x, a.y);
  o.p[1] = cvt2(a.z, a.w);
  o.p[2] = cvt2(b.x, b.y);
  o.p[3] = cvt2(b.z, b.w);
  *(uint4*)(dst + base) = o.u;
}

// ---------- kernel 1: fused Q/K/V projection (blockIdx.y selects which) ----------
// Wave computes a 32x64 strip: 2 A + 4 B loads -> 8 WMMAs per K-step, with
// double-buffered fragment loads. Q scaled by 1/sqrt(Dk); V stored transposed.

__global__ __launch_bounds__(256) void k_proj_qkv(
    const bf16* __restrict__ Xbf, const bf16* __restrict__ Wbf,
    bf16* __restrict__ Qh, bf16* __restrict__ Kh, bf16* __restrict__ Vt) {
  const int p = blockIdx.y;
  const bf16* X = Xbf + (size_t)p * BS * D_MODEL;
  const bf16* W = Wbf + (size_t)p * D_MODEL * D_MODEL;

  const int wave = threadIdx.x >> 5;
  const int lane = threadIdx.x & 31;
  const int row  = lane & 15;
  const int kg   = lane >> 4;

  const int strip = blockIdx.x * 8 + wave;   // 1536 strips of 32x64
  const int tm = strip / (D_MODEL / 64);     // 0..127
  const int tn = strip % (D_MODEL / 64);     // 0..11  (== head index)

  const bf16* xr = X + (size_t)(tm * 32 + row) * D_MODEL + kg * 16;
  const bf16* wr = W + (size_t)(tn * 64 + row) * D_MODEL + kg * 16;

  v8f acc[2][4] = {};

  v16bf a0 = ld16(xr), a1 = ld16(xr + 16 * D_MODEL);
  v16bf b0 = ld16(wr), b1 = ld16(wr + 16 * D_MODEL);
  v16bf b2 = ld16(wr + 32 * D_MODEL), b3 = ld16(wr + 48 * D_MODEL);

#pragma unroll 2
  for (int k0 = 0; k0 < D_MODEL; k0 += 32) {
    const int kn = (k0 + 32 < D_MODEL) ? k0 + 32 : 0;  // harmless wrap on last iter
    v16bf a0n = ld16(xr + kn);
    v16bf a1n = ld16(xr + 16 * D_MODEL + kn);
    v16bf b0n = ld16(wr + kn);
    v16bf b1n = ld16(wr + 16 * D_MODEL + kn);
    v16bf b2n = ld16(wr + 32 * D_MODEL + kn);
    v16bf b3n = ld16(wr + 48 * D_MODEL + kn);

    acc[0][0] = wmma_bf16(a0, b0, acc[0][0]);
    acc[0][1] = wmma_bf16(a0, b1, acc[0][1]);
    acc[0][2] = wmma_bf16(a0, b2, acc[0][2]);
    acc[0][3] = wmma_bf16(a0, b3, acc[0][3]);
    acc[1][0] = wmma_bf16(a1, b0, acc[1][0]);
    acc[1][1] = wmma_bf16(a1, b1, acc[1][1]);
    acc[1][2] = wmma_bf16(a1, b2, acc[1][2]);
    acc[1][3] = wmma_bf16(a1, b3, acc[1][3]);

    a0 = a0n; a1 = a1n;
    b0 = b0n; b1 = b1n; b2 = b2n; b3 = b3n;
  }

  const int h = tn;
#pragma unroll
  for (int ai = 0; ai < 2; ++ai) {
#pragma unroll
    for (int f = 0; f < 4; ++f) {
#pragma unroll
      for (int r = 0; r < 8; ++r) {
        int m = tm * 32 + ai * 16 + r + 8 * kg;
        int bb = m >> 11, s = m & (SS - 1);
        int d = f * 16 + row;
        float v = acc[ai][f][r];
        size_t hs = ((size_t)bb * NH + h) * SS + s;
        if (p == 0)
          Qh[hs * DK + d] = f2bf(v * 0.125f);
        else if (p == 1)
          Kh[hs * DK + d] = f2bf(v);
        else
          Vt[(((size_t)bb * NH + h) * DK + d) * SS + s] = f2bf(v);
      }
    }
  }
}

// ---------- kernel 2: flash attention with cooperative async K/V staging ----------
// Block (8 waves, 128 query rows) shares one (b,h): each 32-key K/V tile is
// staged into LDS ONCE per block via async global->LDS (double-buffered),
// instead of 8 redundant per-wave global streams. Waves read WMMA fragments
// from LDS; per-wave LDS region re-layouts P for the P.V WMMA.

__global__ __launch_bounds__(256) void k_attn(
    const bf16* __restrict__ Qh, const bf16* __restrict__ Kh,
    const bf16* __restrict__ Vt, bf16* __restrict__ ctx) {
  __shared__ bf16 ldsK[2][KT * DK];   // [key][d]   32x64, 4KB per buffer
  __shared__ bf16 ldsV[2][DK * KT];   // [d][key]   64x32, 4KB per buffer
  __shared__ bf16 plds[8][16 * 48];   // per-wave P transpose, padded stride

  const int bh   = blockIdx.y;        // 0..23 (b*12+h)
  const int tid  = threadIdx.x;
  const int wave = tid >> 5;
  const int lane = tid & 31;
  const int row  = lane & 15;
  const int kg   = lane >> 4;
  const int q0   = (blockIdx.x * 8 + wave) * 16;

  const bf16* Kbh = Kh + (size_t)bh * SS * DK;
  const bf16* Vbh = Vt + (size_t)bh * DK * SS;

  // Q fragments resident in VGPRs
  const bf16* Qb = Qh + ((size_t)bh * SS + q0) * DK;
  v16bf qa0 = ld16(Qb + row * DK + kg * 16);
  v16bf qa1 = ld16(Qb + row * DK + 32 + kg * 16);

  v8f o0 = {}, o1 = {}, o2 = {}, o3 = {};
  float mi[8], li[8];
#pragma unroll
  for (int r = 0; r < 8; ++r) { mi[r] = -1e30f; li[r] = 0.0f; }

  bf16* my = plds[wave];

  // cooperative stage of one 32-key tile: 256 threads x (16B K + 16B V)
  auto stage = [&](int j0, int buf) {
    cp16_g2l(Kbh + (size_t)j0 * DK + tid * 8, &ldsK[buf][tid * 8]);
    int d = tid >> 2, part = tid & 3;
    cp16_g2l(Vbh + (size_t)d * SS + j0 + part * 8, &ldsV[buf][d * KT + part * 8]);
  };

  stage(0, 0);
  int buf = 0;
  for (int j0 = 0; j0 < SS; j0 += KT, buf ^= 1) {
    stage_wait_and_sync();   // staging of `buf` visible block-wide

    // fragments from LDS (ds_load_b128 pairs)
    v16bf kb0 = ld16(&ldsK[buf][row * DK + kg * 16]);
    v16bf kb1 = ld16(&ldsK[buf][row * DK + 32 + kg * 16]);
    v16bf kb2 = ld16(&ldsK[buf][(16 + row) * DK + kg * 16]);
    v16bf kb3 = ld16(&ldsK[buf][(16 + row) * DK + 32 + kg * 16]);
    v16bf vb0 = ld16(&ldsV[buf][(row) * KT + kg * 16]);
    v16bf vb1 = ld16(&ldsV[buf][(16 + row) * KT + kg * 16]);
    v16bf vb2 = ld16(&ldsV[buf][(32 + row) * KT + kg * 16]);
    v16bf vb3 = ld16(&ldsV[buf][(48 + row) * KT + kg * 16]);

    // kick off next tile's staging while we compute on this one
    if (j0 + KT < SS) stage(j0 + KT, buf ^ 1);

    // ---- S = Q . K^T (16 x 32 tile) ----
    v8f s0 = {}, s1 = {};
    s0 = wmma_bf16(qa0, kb0, s0);
    s0 = wmma_bf16(qa1, kb1, s0);
    s1 = wmma_bf16(qa0, kb2, s1);
    s1 = wmma_bf16(qa1, kb3, s1);

    // ---- online softmax over the 32 new columns ----
    float p0[8], p1[8];
#pragma unroll
    for (int r = 0; r < 8; ++r) {
      float rm = rmax16(fmaxf(s0[r], s1[r]));
      float mn = fmaxf(mi[r], rm);
      float sc = __expf(mi[r] - mn);
      p0[r] = __expf(s0[r] - mn);
      p1[r] = __expf(s1[r] - mn);
      float rs = rsum16(p0[r] + p1[r]);
      li[r] = li[r] * sc + rs;
      mi[r] = mn;
      o0[r] *= sc; o1[r] *= sc; o2[r] *= sc; o3[r] *= sc;
    }

    // ---- P: C-fragment layout -> A-fragment layout via per-wave LDS ----
#pragma unroll
    for (int r = 0; r < 8; ++r) {
      int rr = r + 8 * kg;
      my[rr * 48 + row]      = f2bf(p0[r]);
      my[rr * 48 + 16 + row] = f2bf(p1[r]);
    }
    v16bf pa = ld16(my + row * 48 + kg * 16);  // same-wave DS RAW, in-order

    // ---- O += P . V ----
    o0 = wmma_bf16(pa, vb0, o0);
    o1 = wmma_bf16(pa, vb1, o1);
    o2 = wmma_bf16(pa, vb2, o2);
    o3 = wmma_bf16(pa, vb3, o3);
  }

  const int b = bh / NH, h = bh % NH;
#pragma unroll
  for (int r = 0; r < 8; ++r) {
    float inv = 1.0f / li[r];
    int q = q0 + r + 8 * kg;
    size_t base = ((size_t)b * SS + q) * D_MODEL + h * DK;
    ctx[base + 0 + row]  = f2bf(o0[r] * inv);
    ctx[base + 16 + row] = f2bf(o1[r] * inv);
    ctx[base + 32 + row] = f2bf(o2[r] * inv);
    ctx[base + 48 + row] = f2bf(o3[r] * inv);
  }
}

// ---------- kernel 3: output projection ctx(bf16) @ Wo^T(bf16) -> fp32 ----------

__global__ __launch_bounds__(256) void k_proj_o(
    const bf16* __restrict__ ctx, const bf16* __restrict__ Wobf,
    float* __restrict__ out) {
  const int wave = threadIdx.x >> 5;
  const int lane = threadIdx.x & 31;
  const int row  = lane & 15;
  const int kg   = lane >> 4;

  const int strip = blockIdx.x * 8 + wave;   // 1536 strips of 32x64
  const int tm = strip / (D_MODEL / 64);
  const int tn = strip % (D_MODEL / 64);

  const bf16* ar = ctx + (size_t)(tm * 32 + row) * D_MODEL + kg * 16;
  const bf16* wr = Wobf + (size_t)(tn * 64 + row) * D_MODEL + kg * 16;

  v8f acc[2][4] = {};

  v16bf a0 = ld16(ar), a1 = ld16(ar + 16 * D_MODEL);
  v16bf b0 = ld16(wr), b1 = ld16(wr + 16 * D_MODEL);
  v16bf b2 = ld16(wr + 32 * D_MODEL), b3 = ld16(wr + 48 * D_MODEL);

#pragma unroll 2
  for (int k0 = 0; k0 < D_MODEL; k0 += 32) {
    const int kn = (k0 + 32 < D_MODEL) ? k0 + 32 : 0;
    v16bf a0n = ld16(ar + kn);
    v16bf a1n = ld16(ar + 16 * D_MODEL + kn);
    v16bf b0n = ld16(wr + kn);
    v16bf b1n = ld16(wr + 16 * D_MODEL + kn);
    v16bf b2n = ld16(wr + 32 * D_MODEL + kn);
    v16bf b3n = ld16(wr + 48 * D_MODEL + kn);

    acc[0][0] = wmma_bf16(a0, b0, acc[0][0]);
    acc[0][1] = wmma_bf16(a0, b1, acc[0][1]);
    acc[0][2] = wmma_bf16(a0, b2, acc[0][2]);
    acc[0][3] = wmma_bf16(a0, b3, acc[0][3]);
    acc[1][0] = wmma_bf16(a1, b0, acc[1][0]);
    acc[1][1] = wmma_bf16(a1, b1, acc[1][1]);
    acc[1][2] = wmma_bf16(a1, b2, acc[1][2]);
    acc[1][3] = wmma_bf16(a1, b3, acc[1][3]);

    a0 = a0n; a1 = a1n;
    b0 = b0n; b1 = b1n; b2 = b2n; b3 = b3n;
  }

#pragma unroll
  for (int ai = 0; ai < 2; ++ai) {
#pragma unroll
    for (int f = 0; f < 4; ++f) {
#pragma unroll
      for (int r = 0; r < 8; ++r) {
        int m = tm * 32 + ai * 16 + r + 8 * kg;
        int n = tn * 64 + f * 16 + row;
        out[(size_t)m * D_MODEL + n] = acc[ai][f][r];
      }
    }
  }
}

// ---------- launch ----------

extern "C" void kernel_launch(void* const* d_in, const int* in_sizes, int n_in,
                              void* d_out, int out_size, void* d_ws, size_t ws_size,
                              hipStream_t stream) {
  const float* Xq = (const float*)d_in[0];
  const float* Xk = (const float*)d_in[1];
  const float* Xv = (const float*)d_in[2];
  const float* Wq = (const float*)d_in[3];
  const float* Wk = (const float*)d_in[4];
  const float* Wv = (const float*)d_in[5];
  const float* Wo = (const float*)d_in[6];
  float* out = (float*)d_out;

  const size_t EX = (size_t)BS * D_MODEL;       // 3,145,728 elems
  const size_t EW = (size_t)D_MODEL * D_MODEL;  //   589,824 elems

  // workspace layout (bf16): Xbf[3] | Wbf[4] | Qh | Kh | Vt ; ctx aliases Xbf
  bf16* Xbf = (bf16*)d_ws;
  bf16* Wbf = Xbf + 3 * EX;
  bf16* Qh  = Wbf + 4 * EW;
  bf16* Kh  = Qh + EX;
  bf16* Vt  = Kh + EX;
  bf16* ctx = Xbf;  // Xbf is dead after k_proj_qkv

  dim3 blk(256);

  const int gx = (int)(EX / 8 / 256);  // 1536
  const int gw = (int)(EW / 8 / 256);  //  288
  hipLaunchKernelGGL(k_cvt, dim3(gx), blk, 0, stream, Xq, Xbf + 0 * EX, (int)EX);
  hipLaunchKernelGGL(k_cvt, dim3(gx), blk, 0, stream, Xk, Xbf + 1 * EX, (int)EX);
  hipLaunchKernelGGL(k_cvt, dim3(gx), blk, 0, stream, Xv, Xbf + 2 * EX, (int)EX);
  hipLaunchKernelGGL(k_cvt, dim3(gw), blk, 0, stream, Wq, Wbf + 0 * EW, (int)EW);
  hipLaunchKernelGGL(k_cvt, dim3(gw), blk, 0, stream, Wk, Wbf + 1 * EW, (int)EW);
  hipLaunchKernelGGL(k_cvt, dim3(gw), blk, 0, stream, Wv, Wbf + 2 * EW, (int)EW);
  hipLaunchKernelGGL(k_cvt, dim3(gw), blk, 0, stream, Wo, Wbf + 3 * EW, (int)EW);

  const int strips = (BS / 32) * (D_MODEL / 64);  // 1536
  hipLaunchKernelGGL(k_proj_qkv, dim3(strips / 8, 3), blk, 0, stream,
                     Xbf, Wbf, Qh, Kh, Vt);

  hipLaunchKernelGGL(k_attn, dim3(SS / 128, BB * NH), blk, 0, stream,
                     Qh, Kh, Vt, ctx);

  hipLaunchKernelGGL(k_proj_o, dim3(strips / 8, 1), blk, 0, stream,
                     ctx, Wbf + 3 * EW, out);

  (void)in_sizes; (void)n_in; (void)out_size; (void)ws_size;
}
